// FusedLinear_9861244911822
// MI455X (gfx1250) — matrix operand order repaired and checked by reading, not compiled, not run on hardware
//
#include <hip/hip_runtime.h>

#define B_DIM 8192
#define HIN   4096
#define HOUT  4096
#define KEXP  4
#define RRANK 8
#define KR    32

typedef __attribute__((ext_vector_type(16))) __bf16        v16bf;
typedef __attribute__((ext_vector_type(8)))  __bf16        v8bf;
typedef __attribute__((ext_vector_type(8)))  float         v8f;
typedef __attribute__((ext_vector_type(4)))  float         v4f;
typedef __attribute__((ext_vector_type(4)))  unsigned int  v4u;

// f32 -> bf16 via native conversion (clang emits hw cvt, RNE)
__device__ __forceinline__ unsigned short bf_bits(float a) {
  union { __bf16 b; unsigned short u; } c;
  c.b = (__bf16)a;
  return c.u;
}

// ---------------------------------------------------------------------------
// Kernel 1: aw[b, kr] = topk[b, kr>>3] * dot(x[b,:], qa[kr,:])   (bf16 out)
// qa is [K, R, HIN] contiguous == [KR, HIN] with kr = k*R + r.
// Block = 32 batch rows x all 32 kr outputs; 256 threads, LDS-tiled over HIN.
// ---------------------------------------------------------------------------
__global__ __launch_bounds__(256) void prep_aw_kernel(
    const float* __restrict__ X, const float* __restrict__ TKW,
    const float* __restrict__ QA, unsigned short* __restrict__ AWo)
{
  __shared__ float Xs[32][68];   // 68 floats/row -> 16B aligned rows
  __shared__ float Qs[32][68];
  const int tid  = threadIdx.x;
  const int b0   = blockIdx.x * 32;
  const int bl   = tid & 31;     // local batch row for compute
  const int kg   = tid >> 5;     // kr group (each owns kr = 4*kg .. 4*kg+3)
  const int lrow = tid >> 3;     // load mapping: 32 rows x 64 cols
  const int lcol = (tid & 7) * 8;

  float a0 = 0.f, a1 = 0.f, a2 = 0.f, a3 = 0.f;

  for (int h0 = 0; h0 < HIN; h0 += 64) {
    v4f x0 = *(const v4f*)(X + (size_t)(b0 + lrow) * HIN + h0 + lcol);
    v4f x1 = *(const v4f*)(X + (size_t)(b0 + lrow) * HIN + h0 + lcol + 4);
    v4f q0 = *(const v4f*)(QA + (size_t)lrow * HIN + h0 + lcol);
    v4f q1 = *(const v4f*)(QA + (size_t)lrow * HIN + h0 + lcol + 4);
    *(v4f*)&Xs[lrow][lcol]     = x0;
    *(v4f*)&Xs[lrow][lcol + 4] = x1;
    *(v4f*)&Qs[lrow][lcol]     = q0;
    *(v4f*)&Qs[lrow][lcol + 4] = q1;
    __syncthreads();
#pragma unroll 8
    for (int c = 0; c < 64; c++) {
      float xv = Xs[bl][c];
      a0 += xv * Qs[kg * 4 + 0][c];
      a1 += xv * Qs[kg * 4 + 1][c];
      a2 += xv * Qs[kg * 4 + 2][c];
      a3 += xv * Qs[kg * 4 + 3][c];
    }
    __syncthreads();
  }

  const int b = b0 + bl;
  unsigned short* op = AWo + (size_t)b * KR + kg * 4;
  op[0] = bf_bits(a0 * TKW[b * KEXP + ((kg * 4 + 0) >> 3)]);
  op[1] = bf_bits(a1 * TKW[b * KEXP + ((kg * 4 + 1) >> 3)]);
  op[2] = bf_bits(a2 * TKW[b * KEXP + ((kg * 4 + 2) >> 3)]);
  op[3] = bf_bits(a3 * TKW[b * KEXP + ((kg * 4 + 3) >> 3)]);
}

// ---------------------------------------------------------------------------
// Kernel 2: qbs[o, kr] = scaling[o] * qb[k, o, r]   (bf16 out)
// ---------------------------------------------------------------------------
__global__ __launch_bounds__(256) void prep_qbs_kernel(
    const float* __restrict__ QB, const float* __restrict__ SC,
    unsigned short* __restrict__ out)
{
  const int idx = blockIdx.x * 256 + threadIdx.x;   // < HOUT*KR
  const int o = idx >> 5, kr = idx & 31;
  const int k = kr >> 3,  r  = kr & 7;
  out[idx] = bf_bits(SC[o] * QB[((size_t)k * HOUT + o) * RRANK + r]);
}

// ---------------------------------------------------------------------------
// Kernel 3: C[256x256 tiles] = Xbf @ Wbf^T  (+ epilogue: AW @ QBS^T)
// 8 waves: waveM = wid&3 (M offset *64), waveN = wid>>2 (N offset *128).
// Each wave: 4x8 grid of 16x16 f32 accumulators, K-steps of 32 (bf16 WMMA).
// LDS: double-buffered Xs/Ws, 256 rows x 32 bf16 each (64 KB dynamic).
// ---------------------------------------------------------------------------
#define TILE_ROWS 256
#define LDS_TILE  (TILE_ROWS * 32)    // shorts per (matrix, buffer)

__global__ __launch_bounds__(256) void fused_gemm_wmma(
    const float* __restrict__ X, const float* __restrict__ W,
    const unsigned short* __restrict__ AW, const unsigned short* __restrict__ QBS,
    float* __restrict__ C)
{
  extern __shared__ unsigned short lds[];
  unsigned short* Xs0 = lds;                 // [2][256][32]
  unsigned short* Ws0 = lds + 2 * LDS_TILE;  // [2][256][32]

  const int tid   = threadIdx.x;
  const int lane  = tid & 31;
  const int wid   = tid >> 5;
  const int waveM = wid & 3;
  const int waveN = wid >> 2;
  const int bm = blockIdx.y, bn = blockIdx.x;
  const int hi  = (lane >> 4) & 1;   // lane half selects K sub-chunks
  const int l16 = lane & 15;

  const float* gX = X + (size_t)(bm * TILE_ROWS + tid) * HIN;
  const float* gW = W + (size_t)(bn * TILE_ROWS + tid) * HIN;
  unsigned short* sXrow = Xs0 + tid * 32;
  unsigned short* sWrow = Ws0 + tid * 32;

  v8f acc[4][8];
  const v8f vzero = {0.f, 0.f, 0.f, 0.f, 0.f, 0.f, 0.f, 0.f};
#pragma unroll
  for (int i = 0; i < 4; i++)
#pragma unroll
    for (int j = 0; j < 8; j++) acc[i][j] = vzero;

  // load 32 contiguous f32 (one tile-row segment) into regs
  auto load_row = [&](const float* g, v4f* r) {
#pragma unroll
    for (int i = 0; i < 8; i++) r[i] = ((const v4f*)g)[i];
  };
  // 32 f32 -> 32 bf16 (native cvt) and store one LDS row (4 x b128)
  auto cvt_store_row = [&](const v4f* r, unsigned short* s) {
#pragma unroll
    for (int i = 0; i < 4; i++) {
      v8bf o;
      o[0] = (__bf16)r[2 * i][0];
      o[1] = (__bf16)r[2 * i][1];
      o[2] = (__bf16)r[2 * i][2];
      o[3] = (__bf16)r[2 * i][3];
      o[4] = (__bf16)r[2 * i + 1][0];
      o[5] = (__bf16)r[2 * i + 1][1];
      o[6] = (__bf16)r[2 * i + 1][2];
      o[7] = (__bf16)r[2 * i + 1][3];
      *(v8bf*)(s + i * 8) = o;
    }
  };

  // one K=32 step: gather A/B fragments per ISA lane layout, 32 WMMAs
  auto compute_step = [&](const unsigned short* Xb, const unsigned short* Wb) {
    v16bf af[4], bfr[8];
#pragma unroll
    for (int i = 0; i < 4; i++) {
      // A 16x32 bf16: lanes0-15 -> K{0-7,16-23}, lanes16-31 -> K{8-15,24-31}
      const unsigned short* p = Xb + (waveM * 64 + i * 16 + l16) * 32 + hi * 8;
      union { v4u u[2]; v16bf v; } t;
      t.u[0] = *(const v4u*)p;
      t.u[1] = *(const v4u*)(p + 16);
      af[i] = t.v;
    }
#pragma unroll
    for (int j = 0; j < 8; j++) {
      // B 32x16 bf16: lane -> column (row of W tile); lanes0-15 K{0-7,16-23}, 16-31 K{8-15,24-31}
      const unsigned short* p = Wb + (waveN * 128 + j * 16 + l16) * 32 + hi * 8;
      union { v4u u[2]; v16bf v; } t;
      t.u[0] = *(const v4u*)p;
      t.u[1] = *(const v4u*)(p + 16);
      bfr[j] = t.v;
    }
#pragma unroll
    for (int i = 0; i < 4; i++)
#pragma unroll
      for (int j = 0; j < 8; j++)
        acc[i][j] = __builtin_amdgcn_wmma_f32_16x16x32_bf16(
            false, af[i], false, bfr[j], (short)0, acc[i][j], false, false);
  };

  v4f rx[8], rw[8];
  load_row(gX, rx);
  load_row(gW, rw);
  cvt_store_row(rx, sXrow);
  cvt_store_row(rw, sWrow);
  __syncthreads();

  const int NK = HIN / 32;   // 128
  for (int ks = 0; ks < NK; ks++) {
    const int cur = ks & 1, nxt = cur ^ 1;
    if (ks + 3 < NK) {
      // prime L2/WGP$ ahead of demand loads (global_prefetch_b8)
      __builtin_prefetch(gX + (size_t)(ks + 3) * 32, 0, 0);
      __builtin_prefetch(gW + (size_t)(ks + 3) * 32, 0, 0);
    }
    if (ks + 1 < NK) {
      load_row(gX + (size_t)(ks + 1) * 32, rx);
      load_row(gW + (size_t)(ks + 1) * 32, rw);
      cvt_store_row(rx, sXrow + nxt * LDS_TILE);
      cvt_store_row(rw, sWrow + nxt * LDS_TILE);
    }
    compute_step(Xs0 + cur * LDS_TILE, Ws0 + cur * LDS_TILE);
    __syncthreads();
  }

  // Epilogue: LoRA delta as one more K=32 step. AW/QBS are bf16, tile shape
  // matches exactly: stage into buffer 0 and reuse the same WMMA path.
  {
    const v4u* ga = (const v4u*)(AW  + (size_t)(bm * TILE_ROWS + tid) * KR);
    const v4u* gq = (const v4u*)(QBS + (size_t)(bn * TILE_ROWS + tid) * KR);
    v4u a0 = ga[0], a1 = ga[1], a2 = ga[2], a3 = ga[3];
    v4u q0 = gq[0], q1 = gq[1], q2 = gq[2], q3 = gq[3];
    ((v4u*)sXrow)[0] = a0; ((v4u*)sXrow)[1] = a1;
    ((v4u*)sXrow)[2] = a2; ((v4u*)sXrow)[3] = a3;
    ((v4u*)sWrow)[0] = q0; ((v4u*)sWrow)[1] = q1;
    ((v4u*)sWrow)[2] = q2; ((v4u*)sWrow)[3] = q3;
    __syncthreads();
    compute_step(Xs0, Ws0);
  }

  // Store: D layout — VGPR v holds M = v + 8*(lane>=16), N = lane%16.
#pragma unroll
  for (int i = 0; i < 4; i++) {
    const int row0 = bm * TILE_ROWS + waveM * 64 + i * 16 + hi * 8;
#pragma unroll
    for (int j = 0; j < 8; j++) {
      const int col = bn * TILE_ROWS + waveN * 128 + j * 16 + l16;
      float* cp = C + (size_t)row0 * HOUT + col;
#pragma unroll
      for (int v = 0; v < 8; v++)
        cp[(size_t)v * HOUT] = acc[i][j][v];
    }
  }
}

// ---------------------------------------------------------------------------
extern "C" void kernel_launch(void* const* d_in, const int* in_sizes, int n_in,
                              void* d_out, int out_size, void* d_ws, size_t ws_size,
                              hipStream_t stream) {
  (void)in_sizes; (void)n_in; (void)out_size; (void)ws_size;
  const float* x   = (const float*)d_in[0];
  const float* tkw = (const float*)d_in[1];
  const float* W   = (const float*)d_in[2];
  const float* qa  = (const float*)d_in[3];
  const float* qb  = (const float*)d_in[4];
  const float* sc  = (const float*)d_in[5];
  float* out = (float*)d_out;

  unsigned short* aw  = (unsigned short*)d_ws;          // B*32 bf16 = 512 KB
  unsigned short* qbs = aw + (size_t)B_DIM * KR;        // HOUT*32 bf16 = 256 KB

  prep_aw_kernel<<<B_DIM / 32, 256, 0, stream>>>(x, tkw, qa, aw);
  prep_qbs_kernel<<<(HOUT * KR) / 256, 256, 0, stream>>>(qb, sc, qbs);

  dim3 grid(HOUT / TILE_ROWS, B_DIM / TILE_ROWS);       // (16, 32)
  fused_gemm_wmma<<<grid, 256, 4 * LDS_TILE * sizeof(unsigned short), stream>>>(
      x, W, aw, qbs, out);
}
